// SIREN_16501264351966
// MI455X (gfx1250) — matrix operand order repaired and checked by reading, not compiled, not run on hardware
//
#include <hip/hip_runtime.h>

typedef __attribute__((ext_vector_type(8)))  _Float16 v8h;
typedef __attribute__((ext_vector_type(16))) _Float16 v16h;
typedef __attribute__((ext_vector_type(8)))  float    v8f;
typedef __attribute__((ext_vector_type(4)))  int      v4i;

#define SIREN_B  32
#define SIREN_N  32768
#define TP       50307
#define BLK_N    256               // points per workgroup
#define NTHREADS 512               // 16 waves; one 16-row M-slab per wave
#define PITCH    136               // 128+8 halfs: 272B rows, 68-dword stride (4 mod 64 banks), 16B aligned
#define OMEGA0   30.0f
#define WSZ      16384             // 128*128 weights per hidden layer
#define WS_HALFS ((size_t)SIREN_B * 3 * WSZ)

#define AS1 __attribute__((address_space(1)))
#define AS3 __attribute__((address_space(3)))

#if defined(__gfx1250__) && __has_builtin(__builtin_amdgcn_global_load_async_to_lds_b128)
#define SIREN_ASYNC 1
#else
#define SIREN_ASYNC 0
#endif

static __device__ __forceinline__ v16h mk16(v8h lo, v8h hi) {
  v16h r;
#pragma unroll
  for (int i = 0; i < 8; ++i) { r[i] = lo[i]; r[i + 8] = hi[i]; }
  return r;
}

// ---- pre-pass: convert the 3 hidden-layer weight matrices of every batch elem to f16 ----
__global__ __launch_bounds__(256)
void siren_w16_prepass(const float* __restrict__ params_all, _Float16* __restrict__ w16) {
  int idx = blockIdx.x * 256 + threadIdx.x;          // grid sized to exactly WS_HALFS
  int b = idx / (3 * WSZ);
  int r = idx - b * (3 * WSZ);
  int l = r >> 14;                                   // / WSZ
  int e = r & (WSZ - 1);
  w16[idx] = (_Float16)params_all[(size_t)b * TP + 384 + l * 16512 + e];
}

template <bool WS16>
__global__ __launch_bounds__(NTHREADS)
void siren_fused_kernel(const float* __restrict__ coords,
                        const float* __restrict__ params_all,
                        const _Float16* __restrict__ w16,
                        float* __restrict__ out) {
  __shared__ _Float16 Xs[BLK_N][PITCH];  // activations, f16
  __shared__ _Float16 Ws[128][PITCH];    // current hidden-layer weights, f16, Ws[out][in]
  __shared__ float    bs[128];
  __shared__ float    xy[BLK_N][2];
  __shared__ float    w0s[384];          // layer0 W (256) + b (128)

  const int t    = threadIdx.x;
  const int wave = t >> 5;
  const int lane = t & 31;
  const int hi   = lane >> 4;
  const int ln   = lane & 15;
  const int tile = blockIdx.x;
  const int b    = blockIdx.y;

  const float* params = params_all + (size_t)b * TP;
  const int p0 = tile * BLK_N;

  // ---- stage coords + layer-0 params ----
  {
    const float* cb = coords + ((size_t)b * SIREN_N + p0) * 2;
    xy[t >> 1][t & 1] = cb[t];                       // 512 floats coalesced
    if (t < 384) w0s[t] = params[t];
  }
  __syncthreads();

  // ---- layer 0: [BLK_N x 2] x [2 x 128] + bias, sin ----
  for (int idx = t; idx < BLK_N * 128; idx += NTHREADS) {
    int p = idx >> 7, o = idx & 127;
    float v = fmaf(xy[p][0], w0s[2 * o],
              fmaf(xy[p][1], w0s[2 * o + 1], w0s[256 + o]));
    Xs[p][o] = (_Float16)__sinf(OMEGA0 * v);
  }

  // ---- hidden layers 1..3: 128x128 GEMMs on WMMA ----
  for (int l = 0; l < 3; ++l) {
    const int boff = 16768 + l * 16512;              // 16768, 33280, 49792

    if constexpr (WS16) {
      const _Float16* wsrc = w16 + ((size_t)b * 3 + l) * WSZ;
#if SIREN_ASYNC
      for (int c = t; c < 2048; c += NTHREADS) {     // 2048 chunks of 16B
        int o = c >> 4, seg = c & 15;
        __builtin_amdgcn_global_load_async_to_lds_b128(
            (AS1 v4i*)(wsrc + c * 8),
            (AS3 v4i*)&Ws[o][seg * 8], 0, 0);
      }
#if __has_builtin(__builtin_amdgcn_s_wait_asynccnt)
      __builtin_amdgcn_s_wait_asynccnt(0);
#else
      asm volatile("s_wait_asynccnt 0x0" ::: "memory");
#endif
#else
      for (int c = t; c < 2048; c += NTHREADS) {
        int o = c >> 4, seg = c & 15;
        *(v8h*)&Ws[o][seg * 8] = *(const v8h*)(wsrc + c * 8);
      }
#endif
    } else {
      const float* wsrc = params + 384 + l * 16512;
      for (int idx = t; idx < WSZ; idx += NTHREADS)
        Ws[idx >> 7][idx & 127] = (_Float16)wsrc[idx];
    }
    if (t < 128) bs[t] = params[boff + t];
    __syncthreads();              // Ws/bs ready; prev-layer Xs writes visible

    v8f c[8] = {};
    const int p = wave * 16 + ln;                    // A-matrix row for this lane

#pragma unroll
    for (int ks = 0; ks < 4; ++ks) {
      // A fragment (16x32 f16): elems 0-7 -> K=32ks+8hi+0..7, elems 8-15 -> +16
      v8h a_lo = *(const v8h*)&Xs[p][ks * 32 + 8 * hi];
      v8h a_hi = *(const v8h*)&Xs[p][ks * 32 + 16 + 8 * hi];
      v16h a = mk16(a_lo, a_hi);
#pragma unroll
      for (int nt = 0; nt < 8; ++nt) {
        // B fragment (32x16 f16): lane col n, elems 0-7 -> K=32ks+8hi+0..7, elems 8-15 -> +16; B[k][n]=W[n][k]
        int n = nt * 16 + ln;
        v8h b_lo = *(const v8h*)&Ws[n][ks * 32 + 8 * hi];
        v8h b_hi = *(const v8h*)&Ws[n][ks * 32 + 16 + 8 * hi];
        v16h bm = mk16(b_lo, b_hi);
        c[nt] = __builtin_amdgcn_wmma_f32_16x16x32_f16(
            false, a, false, bm, (short)0, c[nt], false, false);
      }
    }
    __syncthreads();              // all LDS reads of this layer done

    // bias + sin(30x), re-stripe D layout (col-per-lane) -> A layout (row-per-lane)
#pragma unroll
    for (int nt = 0; nt < 8; ++nt) {
      int n = nt * 16 + ln;
      float bias = bs[n];
#pragma unroll
      for (int j = 0; j < 8; ++j) {
        int pr = wave * 16 + 8 * hi + j;             // D row M = j + 8*hi
        Xs[pr][n] = (_Float16)__sinf(OMEGA0 * (c[nt][j] + bias));
      }
    }
  }
  __syncthreads();

  // ---- output layer: [BLK_N x 128] x [128 x 3] + bias (no sine) ----
  for (int idx = t; idx < BLK_N * 3; idx += NTHREADS) {
    int p = idx / 3, ch = idx - 3 * p;
    const float* wr = params + 49920 + ch * 128;
    float acc = params[50304 + ch];
#pragma unroll 2
    for (int k = 0; k < 128; k += 8) {
      v8h xv = *(const v8h*)&Xs[p][k];
#pragma unroll
      for (int u = 0; u < 8; ++u)
        acc = fmaf((float)xv[u], wr[k + u], acc);
    }
    out[((size_t)b * SIREN_N + p0 + p) * 3 + ch] = acc;
  }
}

extern "C" void kernel_launch(void* const* d_in, const int* in_sizes, int n_in,
                              void* d_out, int out_size, void* d_ws, size_t ws_size,
                              hipStream_t stream) {
  const float* coords = (const float*)d_in[0];   // [B, N, 2] f32
  const float* params = (const float*)d_in[1];   // [B, 50307] f32
  float* out = (float*)d_out;                    // [B, N, 3] f32
  dim3 grid(SIREN_N / BLK_N, SIREN_B, 1);

  if (ws_size >= WS_HALFS * sizeof(_Float16)) {
    _Float16* w16 = (_Float16*)d_ws;
    siren_w16_prepass<<<dim3((unsigned)(WS_HALFS / 256), 1, 1), dim3(256, 1, 1), 0, stream>>>(params, w16);
    siren_fused_kernel<true><<<grid, dim3(NTHREADS, 1, 1), 0, stream>>>(coords, params, w16, out);
  } else {
    siren_fused_kernel<false><<<grid, dim3(NTHREADS, 1, 1), 0, stream>>>(coords, params, nullptr, out);
  }
  (void)in_sizes; (void)n_in; (void)out_size;
}
